// GCNSampling_18141941859038
// MI455X (gfx1250) — matrix operand order repaired and checked by reading, not compiled, not run on hardware
//
#include <hip/hip_runtime.h>
#include <cstdint>
#include <cstddef>

// ---------------------------------------------------------------------------
// GCN pipeline for gfx1250 (MI455X), bandwidth-first design:
//   per layer: degree count -> exclusive scan -> edge binning (counting sort)
//              -> wave-per-node gather mean (pure streaming, no f32 atomics)
//              -> fused WMMA bf16 GEMM + bias (+ReLU)
//   WMMA: v_wmma_f32_16x16x32_bf16 (A/B bf16, C/D f32), wave32 16x16 tiles
// ---------------------------------------------------------------------------

typedef __attribute__((ext_vector_type(16))) __bf16 v16bf;
typedef __attribute__((ext_vector_type(8)))  float  v8f;
typedef __attribute__((ext_vector_type(4)))  float  f4;

union BfVec {
    v16bf  v;
    __bf16 h[16];
    f4     f[2];
};

// ---------------------------------------------------------------- zero ints
__global__ void zero_int(int* __restrict__ p, int n) {
    int i = blockIdx.x * blockDim.x + threadIdx.x;
    if (i < n) p[i] = 0;
}

// ------------------------------------------------- pack weights to WMMA B tile
// Wb[(kt*Npad + n)*32 + kl] = bf16(W[(kt*32+kl)*N + n])  (zero-padded cols)
__global__ void pack_w(const float* __restrict__ W, __bf16* __restrict__ Wb,
                       int K, int N, int Npad) {
    int i     = blockIdx.x * blockDim.x + threadIdx.x;
    int total = (K >> 5) * Npad * 32;
    if (i >= total) return;
    int kl = i & 31;
    int n  = (i >> 5) % Npad;
    int kt = (i >> 5) / Npad;
    float v = (n < N) ? W[(kt * 32 + kl) * N + n] : 0.0f;
    Wb[i] = (__bf16)v;
}

// ----------------------------------------------------------- degree counting
__global__ void count_deg(const int* __restrict__ dst, int E,
                          int* __restrict__ deg) {
    int e = blockIdx.x * blockDim.x + threadIdx.x;
    if (e < E) atomicAdd(&deg[dst[e]], 1);
}

// ----------------------------------------------- scan step 1: per-block sums
__global__ void block_sum(const int* __restrict__ deg, int n,
                          int* __restrict__ bsum) {
    __shared__ int lds[256];
    int i = blockIdx.x * 256 + threadIdx.x;
    lds[threadIdx.x] = (i < n) ? deg[i] : 0;
    __syncthreads();
    for (int s = 128; s > 0; s >>= 1) {
        if (threadIdx.x < s) lds[threadIdx.x] += lds[threadIdx.x + s];
        __syncthreads();
    }
    if (threadIdx.x == 0) bsum[blockIdx.x] = lds[0];
}

// ------------------------------- scan step 2: exclusive scan of <=256 bsums
__global__ void scan_bsum(int* __restrict__ bsum, int nb) {
    __shared__ int lds[256];
    int t   = threadIdx.x;
    int val = (t < nb) ? bsum[t] : 0;
    lds[t] = val;
    __syncthreads();
    for (int ofs = 1; ofs < 256; ofs <<= 1) {
        int v = (t >= ofs) ? lds[t - ofs] : 0;
        __syncthreads();
        lds[t] += v;
        __syncthreads();
    }
    if (t < nb) bsum[t] = lds[t] - val;   // exclusive
}

// ---------------- scan step 3: per-block exclusive scan + rowstart/cursor out
__global__ void block_scan_write(const int* __restrict__ deg, int n,
                                 const int* __restrict__ bofs,
                                 int* __restrict__ rowstart,
                                 int* __restrict__ cursor, int E) {
    __shared__ int lds[256];
    int i   = blockIdx.x * 256 + threadIdx.x;
    int val = (i < n) ? deg[i] : 0;
    lds[threadIdx.x] = val;
    __syncthreads();
    for (int ofs = 1; ofs < 256; ofs <<= 1) {
        int v = (threadIdx.x >= ofs) ? lds[threadIdx.x - ofs] : 0;
        __syncthreads();
        lds[threadIdx.x] += v;
        __syncthreads();
    }
    if (i < n) {
        int excl = bofs[blockIdx.x] + lds[threadIdx.x] - val;
        rowstart[i] = excl;
        cursor[i]   = excl;
        if (i == n - 1) rowstart[n] = E;
    }
}

// ------------------------------------------- counting-sort edges into buckets
__global__ void fill_bins(const int* __restrict__ src,
                          const int* __restrict__ dst, int E,
                          int* __restrict__ cursor, int* __restrict__ sorted) {
    int e = blockIdx.x * blockDim.x + threadIdx.x;
    if (e < E) {
        int pos = atomicAdd(&cursor[dst[e]], 1);
        sorted[pos] = src[e];
    }
}

// ---------------------------------------- wave-per-node gather mean (no atomics)
// F floats per node; lane l owns f4 chunks {l, l+32, ...}: coalesced b128 loads.
template <int F>
__global__ void gather_mean(const float* __restrict__ H,
                            const int* __restrict__ sorted,
                            const int* __restrict__ rowstart,
                            int Nnode, float* __restrict__ outAcc) {
    int node = blockIdx.x * (blockDim.x >> 5) + (threadIdx.x >> 5);
    if (node >= Nnode) return;
    int lane = threadIdx.x & 31;
    int beg = rowstart[node], end = rowstart[node + 1];
    constexpr int C = F / 128;          // f4 chunks per lane (2 for 256, 1 for 128)
    f4 acc[C] = {};
    for (int i = beg; i < end; ++i) {
        int s = sorted[i];
        const f4* p = (const f4*)(H + (size_t)s * F);
#pragma unroll
        for (int c = 0; c < C; ++c) acc[c] += p[lane + c * 32];
    }
    float inv = 1.0f / fmaxf((float)(end - beg), 1.0f);
    f4* o = (f4*)(outAcc + (size_t)node * F);
#pragma unroll
    for (int c = 0; c < C; ++c) o[lane + c * 32] = acc[c] * inv;
}

// --------------------------------------------------------- fused GEMM + bias/act
// Block = (Npad/16) waves; blockIdx.x = 16-row M tile; wave = 16-col N tile.
// ISA 16-bit A layout (16x32): lanes 0-15 row M=l: elems[0..7]=K+0..7,
//   [8..15]=K+16..23; lanes 16-31: K+8..15 / K+24..31.
// ISA 16-bit B layout (32x16): lanes 0-15 col N=l hold K+0..7 / K+16..23;
//   lanes 16-31 col N=l-16 hold K+8..15 / K+24..31 (same order as A).
// D: column fixed per lane (N=lane&15); rows i (lanes<16) / 8+i (lanes>=16).
__global__ void gemm_wmma(const float* __restrict__ A,
                          const __bf16* __restrict__ Wb,
                          const float* __restrict__ bias,
                          float* __restrict__ out,
                          int M, int K, int Npad, int Nout, int relu) {
    const int lane  = threadIdx.x & 31;
    const int ntile = threadIdx.x >> 5;
    const int l15   = lane & 15;
    const int hi    = lane >> 4;

    int row    = blockIdx.x * 16 + l15;
    int rclamp = (row < M) ? row : (M - 1);      // keep EXEC all-ones for WMMA
    const float* arow = A + (size_t)rclamp * K;
    int ncol = ntile * 16 + l15;

    v8f c = {};
    const int KT = K >> 5;
    for (int kt = 0; kt < KT; ++kt) {
        BfVec a, b;
        int ka = kt * 32 + hi * 8;
        f4 a0 = *(const f4*)(arow + ka);
        f4 a1 = *(const f4*)(arow + ka + 4);
        f4 a2 = *(const f4*)(arow + ka + 16);
        f4 a3 = *(const f4*)(arow + ka + 20);
#pragma unroll
        for (int j = 0; j < 4; ++j) {
            a.h[j]      = (__bf16)a0[j];
            a.h[4 + j]  = (__bf16)a1[j];
            a.h[8 + j]  = (__bf16)a2[j];
            a.h[12 + j] = (__bf16)a3[j];
        }
        const f4* bp = (const f4*)(Wb + ((size_t)(kt * Npad + ncol) * 32 + hi * 8));
        b.f[0] = bp[0];
        b.f[1] = bp[2];
        // (neg_a, A, neg_b, B, c_mod, C, reuse_a, reuse_b)
        c = __builtin_amdgcn_wmma_f32_16x16x32_bf16(
                false, a.v, false, b.v, (short)0, c, false, false);
    }

    if (ncol < Nout) {
        float bv = bias[ncol];
#pragma unroll
        for (int i = 0; i < 8; ++i) {
            int m = blockIdx.x * 16 + hi * 8 + i;
            if (m < M) {
                float v = c[i] + bv;
                if (relu) v = fmaxf(v, 0.0f);
                out[(size_t)m * Nout + ncol] = v;
            }
        }
    }
}

// ---------------------------------------------------------------------------
extern "C" void kernel_launch(void* const* d_in, const int* in_sizes, int n_in,
                              void* d_out, int out_size, void* d_ws, size_t ws_size,
                              hipStream_t stream) {
    (void)in_sizes; (void)n_in; (void)out_size; (void)ws_size;

    const float* feat = (const float*)d_in[0];
    const int*   src0 = (const int*)d_in[1];
    const int*   dst0 = (const int*)d_in[2];
    const int*   src1 = (const int*)d_in[3];
    const int*   dst1 = (const int*)d_in[4];
    const int*   src2 = (const int*)d_in[5];
    const int*   dst2 = (const int*)d_in[6];
    const float* W1   = (const float*)d_in[7];
    const float* b1   = (const float*)d_in[8];
    const float* W2   = (const float*)d_in[9];
    const float* b2   = (const float*)d_in[10];
    const float* W3   = (const float*)d_in[11];
    const float* b3   = (const float*)d_in[12];

    const int L1 = 60000, L2 = 15000, L3 = 4000;
    const int E0 = 960000, E1 = 240000, E2 = 64000;
    const int FIN = 256, FH = 128, FOUT = 47, FOP = 48;

    // ---- workspace carve-out ----
    char*  wsb = (char*)d_ws;
    size_t off = 0;
    auto carve = [&](size_t bytes) {
        void* p = wsb + off;
        off = (off + bytes + 255) & ~(size_t)255;   // 256B-align next region
        return p;
    };
    float* acc1 = (float*)carve((size_t)L1 * FIN * 4);
    float* acc2 = (float*)carve((size_t)L2 * FH * 4);
    float* acc3 = (float*)carve((size_t)L3 * FH * 4);
    float* out1 = (float*)carve((size_t)L1 * FH * 4);
    float* out2 = (float*)carve((size_t)L2 * FH * 4);
    __bf16* W1b = (__bf16*)carve((size_t)FIN * FH * 2);
    __bf16* W2b = (__bf16*)carve((size_t)FH * FH * 2);
    __bf16* W3b = (__bf16*)carve((size_t)FH * FOP * 2);
    // degree arrays contiguous -> single zero kernel
    int* deg1 = (int*)carve((size_t)(L1 + L2 + L3) * 4);
    int* deg2 = deg1 + L1;
    int* deg3 = deg2 + L2;
    int* row1 = (int*)carve((size_t)(L1 + 1) * 4);
    int* row2 = (int*)carve((size_t)(L2 + 1) * 4);
    int* row3 = (int*)carve((size_t)(L3 + 1) * 4);
    int* cur1 = (int*)carve((size_t)L1 * 4);
    int* cur2 = (int*)carve((size_t)L2 * 4);
    int* cur3 = (int*)carve((size_t)L3 * 4);
    int* srt0 = (int*)carve((size_t)E0 * 4);
    int* srt1 = (int*)carve((size_t)E1 * 4);
    int* srt2 = (int*)carve((size_t)E2 * 4);
    int* bsum = (int*)carve(256 * 4);               // <=235 block sums, reused

    const int NB1 = (L1 + 255) / 256, NB2 = (L2 + 255) / 256, NB3 = (L3 + 255) / 256;

    // ---- init + weight repack ----
    int nz = L1 + L2 + L3;
    zero_int<<<(nz + 255) / 256, 256, 0, stream>>>(deg1, nz);
    pack_w<<<(FIN * FH + 255) / 256, 256, 0, stream>>>(W1, W1b, FIN, FH, FH);
    pack_w<<<(FH * FH + 255) / 256, 256, 0, stream>>>(W2, W2b, FH, FH, FH);
    pack_w<<<(FH * FOP + 255) / 256, 256, 0, stream>>>(W3, W3b, FH, FOUT, FOP);

    // ---- layer 0 -> 1 ----
    count_deg<<<(E0 + 255) / 256, 256, 0, stream>>>(dst0, E0, deg1);
    block_sum<<<NB1, 256, 0, stream>>>(deg1, L1, bsum);
    scan_bsum<<<1, 256, 0, stream>>>(bsum, NB1);
    block_scan_write<<<NB1, 256, 0, stream>>>(deg1, L1, bsum, row1, cur1, E0);
    fill_bins<<<(E0 + 255) / 256, 256, 0, stream>>>(src0, dst0, E0, cur1, srt0);
    gather_mean<256><<<(L1 + 7) / 8, 256, 0, stream>>>(feat, srt0, row1, L1, acc1);
    gemm_wmma<<<L1 / 16, 256, 0, stream>>>(acc1, W1b, b1, out1, L1, FIN, FH, FH, 1);

    // ---- layer 1 -> 2 ----
    count_deg<<<(E1 + 255) / 256, 256, 0, stream>>>(dst1, E1, deg2);
    block_sum<<<NB2, 256, 0, stream>>>(deg2, L2, bsum);
    scan_bsum<<<1, 256, 0, stream>>>(bsum, NB2);
    block_scan_write<<<NB2, 256, 0, stream>>>(deg2, L2, bsum, row2, cur2, E1);
    fill_bins<<<(E1 + 255) / 256, 256, 0, stream>>>(src1, dst1, E1, cur2, srt1);
    gather_mean<128><<<(L2 + 7) / 8, 256, 0, stream>>>(out1, srt1, row2, L2, acc2);
    gemm_wmma<<<(L2 + 15) / 16, 256, 0, stream>>>(acc2, W2b, b2, out2, L2, FH, FH, FH, 1);

    // ---- layer 2 -> 3 ----
    count_deg<<<(E2 + 255) / 256, 256, 0, stream>>>(dst2, E2, deg3);
    block_sum<<<NB3, 256, 0, stream>>>(deg3, L3, bsum);
    scan_bsum<<<1, 256, 0, stream>>>(bsum, NB3);
    block_scan_write<<<NB3, 256, 0, stream>>>(deg3, L3, bsum, row3, cur3, E2);
    fill_bins<<<(E2 + 255) / 256, 256, 0, stream>>>(src2, dst2, E2, cur3, srt2);
    gather_mean<128><<<(L3 + 7) / 8, 256, 0, stream>>>(out2, srt2, row3, L3, acc3);
    gemm_wmma<<<L3 / 16, 96, 0, stream>>>(acc3, W3b, b3, (float*)d_out,
                                          L3, FH, FOP, FOUT, 0);
}
